// HierarchicalTokenizer_54812372632223
// MI455X (gfx1250) — matrix operand order repaired and checked by reading, not compiled, not run on hardware
//
#include <hip/hip_runtime.h>

// ---------------------------------------------------------------------------
// MI455X (gfx1250) implementation of HierarchicalTokenizer forward.
//
// Sizes: B=4, L=128 (M = B*L = 512 tokens), level dims {256,128,64},
// vocabs {1024,512,256}, relation H=256, H/2=128, R=8.
//
// Strategy: bf16 WMMA (v_wmma_f32_16x16x32_bf16) for every GEMM; the pairwise
// relation MLP is fully fused (no [B,L,L,256] tensor ever touches HBM).
// fp32 accumulation / LN / softmax everywhere. In the fused relation kernel
// each wave owns one N-tile and keeps all W2 K-fragments in registers
// (loaded once per block), so WMMA issue is not L2-bandwidth limited.
//
// Input order (harness flattens setup_inputs(); params dict flattened with
// sorted keys, JAX pytree convention):
//   0: features [4,128,256] f32      1: attention_mask [4,128] bool(u8)
//   2..4:   codebooks[0..2]
//   5..10:  decoders[0]  (W1,W2,b1,b2,be1,g1)   11..16: decoders[1]   17..22: decoders[2]
//   23..32: inter[0] (W1,W2,W3,b1,b2,b3,be1,be2,g1,g2)   33..42: inter[1]
//   43..52: intra[0]   53..62: intra[1]   63..72: intra[2]
//   73: proj0 W [256,128]  74: proj0 b  75: proj1 W [128,64]  76: proj1 b
//
// Output (flat, f32): recon0..2 [4,128,256], intra0..2 [4,128,128,8],
// inter0..1 [4,128,128,8], closs (1).
// ---------------------------------------------------------------------------

typedef __attribute__((ext_vector_type(16))) __bf16 v16bf;
typedef __attribute__((ext_vector_type(8)))  float  v8f;

union FragBF { v16bf v; unsigned int u[8]; };

static __device__ __forceinline__ unsigned short f2bf(float f) {
  union { float f; unsigned int u; } c; c.f = f;
  unsigned int u = c.u;
  unsigned int r = u + 0x7FFFu + ((u >> 16) & 1u);   // round-to-nearest-even
  return (unsigned short)(r >> 16);
}

static __device__ __forceinline__ float gelu_f(float x) {
  return 0.5f * x * (1.0f + erff(x * 0.70710678118654752f));
}

// ---------------------------------------------------------------------------
// Nearest-neighbor quantization: one block per token. Writes bf16 code vector
// and per-token squared distance (for deterministic closs reduction).
// ---------------------------------------------------------------------------
__global__ void quantize_kernel(const float* __restrict__ h,
                                const float* __restrict__ cb,
                                unsigned short* __restrict__ qbf,
                                float* __restrict__ dist,
                                int V, int d) {
  const int t = blockIdx.x, tid = threadIdx.x;
  __shared__ float sh[256];
  __shared__ float rb[256];
  __shared__ int   ri[256];
  if (tid < d) sh[tid] = h[(size_t)t * d + tid];
  __syncthreads();
  float best = 3.4e38f; int bi = 0;
  for (int c = tid; c < V; c += 256) {
    const float* row = cb + (size_t)c * d;
    float s = 0.f;
    for (int k = 0; k < d; ++k) { float df = sh[k] - row[k]; s += df * df; }
    if (s < best) { best = s; bi = c; }
  }
  rb[tid] = best; ri[tid] = bi;
  __syncthreads();
  for (int st = 128; st > 0; st >>= 1) {
    if (tid < st) {
      if (rb[tid + st] < rb[tid] ||
          (rb[tid + st] == rb[tid] && ri[tid + st] < ri[tid])) {
        rb[tid] = rb[tid + st]; ri[tid] = ri[tid + st];
      }
    }
    __syncthreads();
  }
  if (tid == 0) dist[t] = rb[0];
  const int w = ri[0];
  if (tid < d) qbf[(size_t)t * d + tid] = f2bf(cb[(size_t)w * d + tid]);
}

// Deterministic commitment-loss reduction (fixed order; no float atomics).
__global__ void closs_kernel(const float* __restrict__ d0,
                             const float* __restrict__ d1,
                             const float* __restrict__ d2,
                             float* __restrict__ out,
                             float s0, float s1, float s2) {
  __shared__ float red[256];
  const int tid = threadIdx.x;
  float a = (d0[tid] + d0[tid + 256]) * s0 +
            (d1[tid] + d1[tid + 256]) * s1 +
            (d2[tid] + d2[tid + 256]) * s2;
  red[tid] = a; __syncthreads();
  for (int st = 128; st > 0; st >>= 1) {
    if (tid < st) red[tid] += red[tid + st];
    __syncthreads();
  }
  if (tid == 0) out[0] = red[0];
}

// ---------------------------------------------------------------------------
// Weight prep: fp32 [K,N] (row slice starting at row0, leading dim ldw)
//   -> bf16 transposed [N,K] so that WMMA B-fragments load as dwords.
// ---------------------------------------------------------------------------
__global__ void transpose_bf16_kernel(const float* __restrict__ W,
                                      unsigned short* __restrict__ out,
                                      int K, int N, int ldw, int row0) {
  int idx = blockIdx.x * blockDim.x + threadIdx.x;
  if (idx >= N * K) return;
  int n = idx / K, k = idx - n * K;
  out[idx] = f2bf(W[(size_t)(row0 + k) * ldw + n]);
}

__global__ void concat_bf16_kernel(const unsigned short* __restrict__ a,
                                   const unsigned short* __restrict__ b,
                                   unsigned short* __restrict__ out,
                                   int M, int da, int db) {
  int idx = blockIdx.x * blockDim.x + threadIdx.x;
  int dt = da + db;
  if (idx >= M * dt) return;
  int r = idx / dt, c = idx - r * dt;
  out[idx] = (c < da) ? a[(size_t)r * da + c] : b[(size_t)r * db + (c - da)];
}

// Row LayerNorm + exact GELU, fp32 in -> bf16 out. N == 256, block == 256.
__global__ void ln_gelu_kernel(const float* __restrict__ x,
                               const float* __restrict__ g,
                               const float* __restrict__ be,
                               unsigned short* __restrict__ out) {
  const int row = blockIdx.x, t = threadIdx.x;
  __shared__ float red[256];
  __shared__ float s_mean, s_rstd;
  float v = x[(size_t)row * 256 + t];
  red[t] = v; __syncthreads();
  for (int st = 128; st > 0; st >>= 1) { if (t < st) red[t] += red[t + st]; __syncthreads(); }
  if (t == 0) s_mean = red[0] * (1.f / 256.f);
  __syncthreads();
  float m = s_mean, dv = v - m;
  red[t] = dv * dv; __syncthreads();
  for (int st = 128; st > 0; st >>= 1) { if (t < st) red[t] += red[t + st]; __syncthreads(); }
  if (t == 0) s_rstd = rsqrtf(red[0] * (1.f / 256.f) + 1e-5f);
  __syncthreads();
  float y = dv * s_rstd * g[t] + be[t];
  out[(size_t)row * 256 + t] = f2bf(gelu_f(y));
}

// ---------------------------------------------------------------------------
// Generic bf16 WMMA GEMM: C[M,N] = A[M,K] @ B[K,N] (+bias), A bf16 row-major,
// B provided transposed bf16 [N,K]. One wave per 16x16 output tile.
// ISA 16-bit A(16x32) layout: lane = m + 16*g; VGPR v holds K pair
//   k = 2v + 8g + (v>=4 ? 8 : 0). B(32x16): lane = n + 16*g, same k order;
//   this code uses k = 2v + 16g for both A and B. C: VGPR r -> (M = r + 8g, N = lane&15).
// ---------------------------------------------------------------------------
__global__ void gemm_bf16_kernel(const unsigned short* __restrict__ A,
                                 const unsigned short* __restrict__ Bt,
                                 const float* __restrict__ bias,
                                 float* __restrict__ C,
                                 int M, int N, int K) {
  const int wave = threadIdx.x >> 5, lane = threadIdx.x & 31;
  const int tiles_m = M >> 4;
  const int tiles = tiles_m * (N >> 4);
  const int tile = blockIdx.x * 4 + wave;
  if (tile >= tiles) return;
  const int mt = tile % tiles_m, nt = tile / tiles_m;
  const int m = lane & 15, g = lane >> 4;
  const unsigned short* ap = A  + (size_t)(mt * 16 + m) * K;
  const unsigned short* bp = Bt + (size_t)(nt * 16 + m) * K;
  v8f acc = {};
  for (int k0 = 0; k0 < K; k0 += 32) {
    FragBF fa, fb;
#pragma unroll
    for (int v2 = 0; v2 < 8; ++v2) {
      const int ka = k0 + 2 * v2 + 16 * g;
      const int kb = k0 + 2 * v2 + 16 * g;
      fa.u[v2] = *(const unsigned int*)(ap + ka);
      fb.u[v2] = *(const unsigned int*)(bp + kb);
    }
    acc = __builtin_amdgcn_wmma_f32_16x16x32_bf16(false, fa.v, false, fb.v,
                                                  (short)0, acc, false, false);
  }
#pragma unroll
  for (int r = 0; r < 8; ++r) {
    const int row = mt * 16 + r + 8 * g;
    const int col = nt * 16 + m;
    const float bv = bias ? bias[col] : 0.f;
    C[(size_t)row * N + col] = acc[r] + bv;
  }
}

// ---------------------------------------------------------------------------
// Fused pairwise relation predictor. One block = 16 i-values x 2 j-values
// (32 pairs), 256 threads = 8 waves. H=256, H2=128, R=8 fixed.
//   h = GELU(LN(a_i + b_j + b1));  h2 = GELU(LN(h @ W2 + b2));
//   out = softmax(h2 @ W3 + b3) * mask_pair
// Phase 2: wave w owns N-tile w; all 8 K-step W2 fragments are loaded into
// registers once per block (64 VGPRs) and reused for both 16-row tiles with
// dual accumulators -> W2 is read exactly once per block, WMMAs issue
// back-to-back against LDS-resident activations.
// ---------------------------------------------------------------------------
#define ACT_STRIDE 260   // bf16 elems; 130 dwords % 64 banks = 2 -> no conflicts
#define H2_STRIDE  132

__global__ void relation_kernel(const float* __restrict__ arow,   // [512,256]
                                const float* __restrict__ brow,   // [512,256]
                                const float* __restrict__ b1,
                                const float* __restrict__ g1,
                                const float* __restrict__ be1,
                                const unsigned short* __restrict__ W2t, // [128,256] bf16
                                const float* __restrict__ b2,
                                const float* __restrict__ g2,
                                const float* __restrict__ be2,
                                const float* __restrict__ W3,     // [128,8]
                                const float* __restrict__ b3,     // [8]
                                const unsigned char* __restrict__ mask, // [512]
                                float* __restrict__ out) {        // [4,128,128,8]
  __shared__ unsigned short act[32][ACT_STRIDE]; // 32 pair-rows x 256 bf16
  __shared__ float h2s[32][H2_STRIDE];           // 32 pair-rows x 128 f32

  const int jt = blockIdx.x & 63;          // L/2  = 64 j-tiles
  const int it = (blockIdx.x >> 6) & 7;    // L/16 = 8  i-tiles
  const int bb = blockIdx.x >> 9;          // batch
  const int wave = threadIdx.x >> 5, lane = threadIdx.x & 31;
  const int m = lane & 15, g = lane >> 4;

  // ---- Load this wave's W2 N-tile fragments (all 8 K-steps) into regs -----
  FragBF fb[8];
  {
    const unsigned short* bp = W2t + (size_t)(wave * 16 + m) * 256;
#pragma unroll
    for (int ks = 0; ks < 8; ++ks)
#pragma unroll
      for (int v2 = 0; v2 < 8; ++v2)
        fb[ks].u[v2] = *(const unsigned int*)(bp + ks * 32 + 2 * v2 + 16 * g);
  }

  // ---- Phase 1: rows = a_i + b_j + b1 -> LN -> GELU -> bf16 in LDS --------
  for (int rr = 0; rr < 4; ++rr) {
    const int p = wave * 4 + rr;
    const int i = it * 16 + (p >> 1);
    const int j = jt * 2 + (p & 1);
    const float* ar = arow + ((size_t)bb * 128 + i) * 256;
    const float* br = brow + ((size_t)bb * 128 + j) * 256;
    float v[8], s1 = 0.f, s2 = 0.f;
#pragma unroll
    for (int t = 0; t < 8; ++t) {
      const int k = t * 32 + lane;
      float x = ar[k] + br[k] + b1[k];
      v[t] = x; s1 += x; s2 += x * x;
    }
    for (int off = 16; off > 0; off >>= 1) {
      s1 += __shfl_xor(s1, off); s2 += __shfl_xor(s2, off);
    }
    const float mean = s1 * (1.f / 256.f);
    const float var  = s2 * (1.f / 256.f) - mean * mean;
    const float rstd = rsqrtf(var + 1e-5f);
#pragma unroll
    for (int t = 0; t < 8; ++t) {
      const int k = t * 32 + lane;
      float y = (v[t] - mean) * rstd * g1[k] + be1[k];
      act[p][k] = f2bf(gelu_f(y));
    }
  }
  __syncthreads();

  // ---- Phase 2: [32,256] @ W2[256,128]; wave = N-tile, B frags in regs ----
  {
    v8f acc0 = {}, acc1 = {};
    for (int ks = 0; ks < 8; ++ks) {
      FragBF fa0, fa1;
#pragma unroll
      for (int v2 = 0; v2 < 8; ++v2) {
        const int ka = ks * 32 + 2 * v2 + 16 * g;
        fa0.u[v2] = *(const unsigned int*)&act[m][ka];
        fa1.u[v2] = *(const unsigned int*)&act[16 + m][ka];
      }
      acc0 = __builtin_amdgcn_wmma_f32_16x16x32_bf16(false, fa0.v, false, fb[ks].v,
                                                     (short)0, acc0, false, false);
      acc1 = __builtin_amdgcn_wmma_f32_16x16x32_bf16(false, fa1.v, false, fb[ks].v,
                                                     (short)0, acc1, false, false);
    }
#pragma unroll
    for (int r = 0; r < 8; ++r) {
      h2s[r + 8 * g][wave * 16 + m]      = acc0[r];
      h2s[16 + r + 8 * g][wave * 16 + m] = acc1[r];
    }
  }
  __syncthreads();

  // ---- Phase 3: +b2 -> LN -> GELU -> @W3 + b3 -> softmax -> mask -> store -
  {
    const int p = wave * 4 + (lane >> 3);   // pair-row, 8 lanes per row
    const int q = lane & 7;
    float xv[16], s1 = 0.f, s2 = 0.f;
#pragma unroll
    for (int t = 0; t < 16; ++t) {
      const int k = q * 16 + t;
      float x = h2s[p][k] + b2[k];
      xv[t] = x; s1 += x; s2 += x * x;
    }
    for (int off = 1; off < 8; off <<= 1) {
      s1 += __shfl_xor(s1, off); s2 += __shfl_xor(s2, off);
    }
    const float mean = s1 * (1.f / 128.f);
    const float var  = s2 * (1.f / 128.f) - mean * mean;
    const float rstd = rsqrtf(var + 1e-5f);
    float part[8] = {0.f, 0.f, 0.f, 0.f, 0.f, 0.f, 0.f, 0.f};
#pragma unroll
    for (int t = 0; t < 16; ++t) {
      const int k = q * 16 + t;
      const float gg = gelu_f((xv[t] - mean) * rstd * g2[k] + be2[k]);
      const float* w3r = W3 + (size_t)k * 8;
#pragma unroll
      for (int r = 0; r < 8; ++r) part[r] += gg * w3r[r];
    }
    for (int off = 1; off < 8; off <<= 1)
#pragma unroll
      for (int r = 0; r < 8; ++r) part[r] += __shfl_xor(part[r], off);

    if (q == 0) {
      const int i = it * 16 + (p >> 1);
      const int j = jt * 2 + (p & 1);
      float mx = -3.4e38f;
#pragma unroll
      for (int r = 0; r < 8; ++r) { part[r] += b3[r]; mx = fmaxf(mx, part[r]); }
      float se = 0.f;
#pragma unroll
      for (int r = 0; r < 8; ++r) { part[r] = expf(part[r] - mx); se += part[r]; }
      const float mval =
          (mask[bb * 128 + i] && mask[bb * 128 + j]) ? (1.f / se) : 0.f;
      const size_t o = (((size_t)bb * 128 + i) * 128 + j) * 8;
#pragma unroll
      for (int r = 0; r < 8; ++r) out[o + r] = part[r] * mval;
    }
  }
}

// ---------------------------------------------------------------------------
// Host orchestration
// ---------------------------------------------------------------------------
static inline int cdiv(long long a, long long b) { return (int)((a + b - 1) / b); }

extern "C" void kernel_launch(void* const* d_in, const int* in_sizes, int n_in,
                              void* d_out, int out_size, void* d_ws, size_t ws_size,
                              hipStream_t stream) {
  (void)in_sizes; (void)n_in; (void)out_size; (void)ws_size;
  const int M = 512;                       // B*L
  const int LD[3] = {256, 128, 64};
  const int VC[3] = {1024, 512, 256};

  const float* features = (const float*)d_in[0];
  const unsigned char* mask = (const unsigned char*)d_in[1];
  const float* cb[3] = {(const float*)d_in[2], (const float*)d_in[3], (const float*)d_in[4]};
  auto F = [&](int i) { return (const float*)d_in[i]; };

  float* out = (float*)d_out;
  const size_t RECON = 0;                         // 3 x 131072
  const size_t REL   = 3 * 131072;                // 5 x 524288 (intra0..2, inter0..1)
  const size_t CLOSS = REL + 5 * 524288;

  // Workspace bump allocator
  char* ws = (char*)d_ws;
  size_t off = 0;
  auto alloc = [&](size_t bytes) -> char* {
    size_t cur = (off + 255) & ~(size_t)255; off = cur + bytes; return ws + cur;
  };
  unsigned short* qbf[3] = {(unsigned short*)alloc((size_t)M * 256 * 2),
                            (unsigned short*)alloc((size_t)M * 128 * 2),
                            (unsigned short*)alloc((size_t)M * 64 * 2)};
  float* h1   = (float*)alloc((size_t)M * 128 * 4);
  float* h2l  = (float*)alloc((size_t)M * 64 * 4);
  float* dist[3] = {(float*)alloc(M * 4), (float*)alloc(M * 4), (float*)alloc(M * 4)};
  float* tmp1 = (float*)alloc((size_t)M * 256 * 4);
  unsigned short* x1bf = (unsigned short*)alloc((size_t)M * 256 * 2);
  unsigned short* cc[2] = {(unsigned short*)alloc((size_t)M * 384 * 2),
                           (unsigned short*)alloc((size_t)M * 192 * 2)};
  unsigned short* wtA = (unsigned short*)alloc((size_t)256 * 384 * 2);
  unsigned short* wtC = (unsigned short*)alloc((size_t)128 * 256 * 2);
  float* arow = (float*)alloc((size_t)M * 256 * 4);
  float* brow = (float*)alloc((size_t)M * 256 * 4);

  auto gemm = [&](const unsigned short* A, const unsigned short* Bt,
                  const float* bias, float* C, int Mm, int N, int K) {
    int tiles = (Mm >> 4) * (N >> 4);
    gemm_bf16_kernel<<<cdiv(tiles, 4), 128, 0, stream>>>(A, Bt, bias, C, Mm, N, K);
  };
  auto xpose = [&](const float* W, unsigned short* o, int K, int N, int ldw, int row0) {
    transpose_bf16_kernel<<<cdiv((long long)N * K, 256), 256, 0, stream>>>(W, o, K, N, ldw, row0);
  };

  // ---- Quantization cascade ----
  quantize_kernel<<<M, 256, 0, stream>>>(features, cb[0], qbf[0], dist[0], VC[0], LD[0]);
  xpose(F(73), wtA, 256, 128, 128, 0);
  gemm(qbf[0], wtA, F(74), h1, M, 128, 256);
  quantize_kernel<<<M, 256, 0, stream>>>(h1, cb[1], qbf[1], dist[1], VC[1], LD[1]);
  xpose(F(75), wtA, 128, 64, 64, 0);
  gemm(qbf[1], wtA, F(76), h2l, M, 64, 128);
  quantize_kernel<<<M, 256, 0, stream>>>(h2l, cb[2], qbf[2], dist[2], VC[2], LD[2]);
  closs_kernel<<<1, 256, 0, stream>>>(dist[0], dist[1], dist[2], out + CLOSS,
                                      1.25f / (512.f * 256.f),
                                      1.25f / (512.f * 128.f),
                                      1.25f / (512.f * 64.f));

  // ---- Decoders: recon_i = GELU(LN(q_i@W1+b1)) @ W2 + b2 ----
  for (int i = 0; i < 3; ++i) {
    const int base = 5 + 6 * i;
    xpose(F(base + 0), wtA, LD[i], 256, 256, 0);
    gemm(qbf[i], wtA, F(base + 2), tmp1, M, 256, LD[i]);
    ln_gelu_kernel<<<M, 256, 0, stream>>>(tmp1, F(base + 5), F(base + 4), x1bf);
    xpose(F(base + 1), wtA, 256, 256, 256, 0);
    gemm(x1bf, wtA, F(base + 3), out + RECON + (size_t)i * 131072, M, 256, 256);
  }

  // ---- Concatenated tokens for inter relations ----
  concat_bf16_kernel<<<cdiv((long long)M * 384, 256), 256, 0, stream>>>(qbf[0], qbf[1], cc[0], M, 256, 128);
  concat_bf16_kernel<<<cdiv((long long)M * 192, 256), 256, 0, stream>>>(qbf[1], qbf[2], cc[1], M, 128, 64);

  // ---- Relation predictors (3 intra + 2 inter) ----
  for (int r = 0; r < 5; ++r) {
    const unsigned short* tok;
    int din, base;
    if (r < 3) { tok = qbf[r]; din = LD[r]; base = 43 + 10 * r; }
    else       { tok = cc[r - 3]; din = LD[r - 3] + LD[r - 2]; base = 23 + 10 * (r - 3); }
    // a = T @ W1[:din], b = T @ W1[din:2din]  (bias b1 folded into phase 1)
    xpose(F(base + 0), wtA, din, 256, 256, 0);
    gemm(tok, wtA, nullptr, arow, M, 256, din);
    xpose(F(base + 0), wtA, din, 256, 256, din);
    gemm(tok, wtA, nullptr, brow, M, 256, din);
    xpose(F(base + 1), wtC, 256, 128, 128, 0);   // W2 -> [128,256] bf16
    relation_kernel<<<4 * 8 * 64, 256, 0, stream>>>(
        arow, brow,
        F(base + 3), F(base + 8), F(base + 6),   // b1, g1, be1
        wtC,
        F(base + 4), F(base + 9), F(base + 7),   // b2, g2, be2
        F(base + 2), F(base + 5),                // W3, b3
        mask, out + REL + (size_t)r * 524288);
  }
}
